// RIMNetwork_68736656605803
// MI455X (gfx1250) — compile-verified, matrix operand order checked by reading, not run on hardware
//
#include <hip/hip_runtime.h>

// ---------------------------------------------------------------------------
// RIM network on gfx1250 (MI455X).  Plan:
//  k1: parallel input projections (x@W_in, then @Wp^T + bp) via bf16 WMMA.
//      Writes A_in (output #3) and stashes input_part into the attn_comm
//      output region (scratch; overwritten with A_comm by k2 at step t).
//  k2: sequential recurrence, 16 batch rows/block (8 blocks, 8 waves = 8
//      mechanisms).  Low-rank rec, Q/K projections via WMMA; attention,
//      tanh, softmax in f32 VALU.  State + transposed bf16 weights in LDS.
//  k3: parallel output projection hiddens@Wo^T via bf16 WMMA.
// ---------------------------------------------------------------------------

typedef __bf16 bf16;
typedef __attribute__((ext_vector_type(16))) __bf16 v16bf;
typedef __attribute__((ext_vector_type(8)))  __bf16 v8bf;
typedef __attribute__((ext_vector_type(8)))  float  v8f;

#define DEVINL __device__ __forceinline__

namespace {
constexpr int kB = 128, kS = 512, kI = 128, kH = 128, kM = 8, kKD = 16, kR = 16, kO = 128;
constexpr long long OUT_OUT   = 0;                       // [B,S,O]
constexpr long long OUT_HID   = 8388608LL;               // [S,B,M,H]
constexpr long long OUT_AIN   = 75497472LL;              // [S,B,M,H]
constexpr long long OUT_ACOMM = 142606336LL;             // [S,B,M,H]
constexpr long long OUT_ATTN  = 209715200LL;             // [S,B,M,M]
}

DEVINL v8f vzero8() {
  v8f z;
#pragma unroll
  for (int i = 0; i < 8; ++i) z[i] = 0.0f;
  return z;
}

DEVINL v8f wmma_bf16(v16bf a, v16bf b, v8f c) {
  // v_wmma_f32_16x16x32_bf16  D = A(16x32) * B(32x16) + C (f32 accum)
  return __builtin_amdgcn_wmma_f32_16x16x32_bf16(false, a, false, b, (short)0, c, false, false);
}

// CDNA5 16-bit operand layout: fragment element j <-> K = koff + (j&7) + (j>=8)*16,
// koff = kk + ((lane>=16) ? 8 : 0).  A: lane&15 = row, B (assumed symmetric): lane&15 = col.
DEVINL v16bf frag_bf16(const bf16* row, int koff) {
  v8bf lo = *reinterpret_cast<const v8bf*>(row + koff);
  v8bf hi = *reinterpret_cast<const v8bf*>(row + koff + 16);
  return __builtin_shufflevector(lo, hi, 0, 1, 2, 3, 4, 5, 6, 7, 8, 9, 10, 11, 12, 13, 14, 15);
}

DEVINL v16bf frag_f32(const float* row, int koff) {
  v16bf r;
#pragma unroll
  for (int i = 0; i < 8; ++i) {
    r[i]     = (bf16)row[koff + i];
    r[i + 8] = (bf16)row[koff + 16 + i];
  }
  return r;
}

// K_real = 16 fragments: upper half (K>=16) zeroed so a K=32 WMMA computes a K=16 product.
DEVINL v16bf fragA_k16_f32(const float* row, int c0) {
  v16bf r;
#pragma unroll
  for (int i = 0; i < 8; ++i) { r[i] = (bf16)row[c0 + i]; r[i + 8] = (bf16)0.0f; }
  return r;
}

DEVINL v16bf fragB_k16_bf16(const bf16* row, int c0) {
  v8bf lo = *reinterpret_cast<const v8bf*>(row + c0);
  v16bf r;
#pragma unroll
  for (int i = 0; i < 8; ++i) { r[i] = lo[i]; r[i + 8] = (bf16)0.0f; }
  return r;
}

// ---------------------------------------------------------------------------
// k1: per block = one (mechanism m, 128-row chunk of the 65536 (b,s) rows).
// LDS: W_in^T bf16 [128][136], Wp bf16 [128][136], A tile f32 [128][132], bp.
// ---------------------------------------------------------------------------
__global__ void __launch_bounds__(256)
k1_input_proj(const float* __restrict__ x, const float* __restrict__ Wp,
              const float* __restrict__ bp, const float* __restrict__ W_in,
              float* __restrict__ ain_out, float* __restrict__ p_out) {
  extern __shared__ __align__(16) char smem[];
  bf16*  Wt  = reinterpret_cast<bf16*>(smem);             // [128][136]  W_in^T
  bf16*  Wpb = reinterpret_cast<bf16*>(smem + 34816);     // [128][136]  Wp rows (already B-layout)
  float* Al  = reinterpret_cast<float*>(smem + 69632);    // [128][132]
  float* bpl = reinterpret_cast<float*>(smem + 137216);   // [128]

  const int tid = threadIdx.x;
  const int m   = blockIdx.x >> 9;          // 512 row-chunks per mechanism
  const int rb0 = (blockIdx.x & 511) << 7;  // first row of this chunk

  const float* Wsrc = W_in + (size_t)m * kI * kH;
#pragma unroll 1
  for (int idx = tid; idx < kI * kH; idx += 256) {
    int i = idx >> 7, h = idx & 127;
    Wt[h * 136 + i] = (bf16)Wsrc[idx];       // transpose: Bt[n=h][k=i]
  }
  const float* Wpsrc = Wp + (size_t)m * kH * kH;
#pragma unroll 1
  for (int idx = tid; idx < kH * kH; idx += 256) {
    int o = idx >> 7, i = idx & 127;
    Wpb[o * 136 + i] = (bf16)Wpsrc[idx];     // Bt[n=o][k=i] = Wp[o][i]
  }
  if (tid < kH) bpl[tid] = bp[m * kH + tid];
  __syncthreads();

  const int w = tid >> 5, lane = tid & 31, ln = lane & 15, hi = lane >> 4, c0 = hi << 3;
  const float* xrow = x + (size_t)(rb0 + w * 16 + ln) * kI;   // rows enumerate b*S+s

  // GEMM1: A_in = x @ W_in[m]
  v8f acc[8];
#pragma unroll
  for (int ct = 0; ct < 8; ++ct) acc[ct] = vzero8();
#pragma unroll
  for (int kk = 0; kk < 128; kk += 32) {
    v16bf a = frag_f32(xrow, kk + c0);
#pragma unroll
    for (int ct = 0; ct < 8; ++ct)
      acc[ct] = wmma_bf16(a, frag_bf16(Wt + (ct * 16 + ln) * 136, kk + c0), acc[ct]);
  }
#pragma unroll
  for (int ct = 0; ct < 8; ++ct)
#pragma unroll
    for (int v = 0; v < 8; ++v)
      Al[(w * 16 + v + 8 * hi) * 132 + ct * 16 + ln] = acc[ct][v];
  __syncthreads();

  // GEMM2: P = A_in @ Wp[m]^T  (A operand re-read from the LDS tile = transpose)
  v8f pacc[8];
#pragma unroll
  for (int ct = 0; ct < 8; ++ct) pacc[ct] = vzero8();
  const float* arow = Al + (w * 16 + ln) * 132;
#pragma unroll
  for (int kk = 0; kk < 128; kk += 32) {
    v16bf a = frag_f32(arow, kk + c0);
#pragma unroll
    for (int ct = 0; ct < 8; ++ct)
      pacc[ct] = wmma_bf16(a, frag_bf16(Wpb + (ct * 16 + ln) * 136, kk + c0), pacc[ct]);
  }

  // coalesced store of A_in tile (reads Al only; runs alongside GEMM2 reads)
#pragma unroll 1
  for (int idx = tid; idx < 128 * 32; idx += 256) {
    int row = idx >> 5, q = idx & 31;
    int rb = rb0 + row, bb = rb >> 9, ss = rb & 511;
    float4 vv = *reinterpret_cast<const float4*>(Al + row * 132 + q * 4);
    *reinterpret_cast<float4*>(ain_out + ((size_t)(ss * kB + bb) * kM + m) * kH + q * 4) = vv;
  }
  __syncthreads();

#pragma unroll
  for (int ct = 0; ct < 8; ++ct) {
    float bb = bpl[ct * 16 + ln];
#pragma unroll
    for (int v = 0; v < 8; ++v)
      Al[(w * 16 + v + 8 * hi) * 132 + ct * 16 + ln] = pacc[ct][v] + bb;
  }
  __syncthreads();
#pragma unroll 1
  for (int idx = tid; idx < 128 * 32; idx += 256) {
    int row = idx >> 5, q = idx & 31;
    int rb = rb0 + row, bb = rb >> 9, ss = rb & 511;
    float4 vv = *reinterpret_cast<const float4*>(Al + row * 132 + q * 4);
    *reinterpret_cast<float4*>(p_out + ((size_t)(ss * kB + bb) * kM + m) * kH + q * 4) = vv;
  }
}

// ---------------------------------------------------------------------------
// k2: recurrence.  grid = 8 blocks x 256 threads; block owns 16 batch rows,
// wave w owns mechanism m = w.  ~260 KB dynamic LDS (WGP has 320 KB).
// ---------------------------------------------------------------------------
__global__ void __launch_bounds__(256)
k2_recurrence(const float* __restrict__ U, const float* __restrict__ V,
              const float* __restrict__ cell_bias, const float* __restrict__ W_Q,
              const float* __restrict__ W_K, const float* __restrict__ W_V,
              float* __restrict__ pc,        // [S,B,M,H] input_part in, A_comm out
              float* __restrict__ hid_out,   // [S,B,M,H]
              float* __restrict__ attn_out)  // [S,B,M,M]
{
  extern __shared__ __align__(16) char smem[];
  float* Hs  = reinterpret_cast<float*>(smem + 0);        // [M*16][132] state (f32)
  bf16*  Ut  = reinterpret_cast<bf16*>(smem + 67584);     // [M*16][136]  U^T
  bf16*  Vt  = reinterpret_cast<bf16*>(smem + 102400);    // [M*128][24]  V^T (K=16 used)
  bf16*  WQl = reinterpret_cast<bf16*>(smem + 151552);    // [M*16][136]  W_Q^T
  bf16*  WKl = reinterpret_cast<bf16*>(smem + 186368);    // [M*16][136]  W_K^T
  float* dVd = reinterpret_cast<float*>(smem + 221184);   // [M][128] diag(W_V)
  float* cb  = reinterpret_cast<float*>(smem + 225280);   // [M][128]
  float* Kc  = reinterpret_cast<float*>(smem + 229376);   // [M*16][16]
  float* Qc  = reinterpret_cast<float*>(smem + 237568);   // [M*16][16]
  float* atl = reinterpret_cast<float*>(smem + 245760);   // [16][8][8]
  float* Sr  = reinterpret_cast<float*>(smem + 249856);   // [M*16][20]

  const int tid = threadIdx.x;
  const int b0  = blockIdx.x * 16;

  // ---- stage transposed bf16 weights / diagonals (once) ----
#pragma unroll 1
  for (int idx = tid; idx < kM * kH * kR; idx += 256) {   // U [M,H,R] -> Ut[m][r][h]
    int r = idx & 15, h = (idx >> 4) & 127, mm = idx >> 11;
    Ut[(mm * 16 + r) * 136 + h] = (bf16)U[idx];
  }
#pragma unroll 1
  for (int idx = tid; idx < kM * kR * kH; idx += 256) {   // V [M,R,H] -> Vt[m][h][r]
    int h = idx & 127, r = (idx >> 7) & 15, mm = idx >> 11;
    Vt[(mm * 128 + h) * 24 + r] = (bf16)V[idx];
  }
#pragma unroll 1
  for (int idx = tid; idx < kM * kH * kKD; idx += 256) {  // [M,H,KD] -> [m][kd][h]
    int kd = idx & 15, h = (idx >> 4) & 127, mm = idx >> 11;
    WQl[(mm * 16 + kd) * 136 + h] = (bf16)W_Q[idx];
    WKl[(mm * 16 + kd) * 136 + h] = (bf16)W_K[idx];
  }
#pragma unroll 1
  for (int idx = tid; idx < kM * kH; idx += 256) {
    int mm = idx >> 7, h = idx & 127;
    dVd[idx] = W_V[((size_t)(mm * kH + h)) * kH + h];     // einsum 'mhh' diagonal
    cb[idx]  = cell_bias[idx];
  }
#pragma unroll 1
  for (int idx = tid; idx < kM * 16 * 132; idx += 256) Hs[idx] = 0.0f;  // H0 = 0
  __syncthreads();

  const int w = tid >> 5, lane = tid & 31, ln = lane & 15, hi = lane >> 4, c0 = hi << 3;
  const int m = w;
  float*      HsRow = Hs  + (m * 16 + ln) * 132;
  const bf16* UtRow = Ut  + (m * 16 + ln) * 136;
  const bf16* WKRow = WKl + (m * 16 + ln) * 136;
  const bf16* WQRow = WQl + (m * 16 + ln) * 136;
  const int rowC = tid >> 4, qC = tid & 15;

#pragma unroll 1
  for (int t = 0; t < kS; ++t) {
    // ---- A1: Srec = Hs_m @ U_m  (16x16, K=128) ----
    v8f sa = vzero8();
#pragma unroll
    for (int kk = 0; kk < 128; kk += 32) {
      v16bf a = frag_f32(HsRow, kk + c0);
      sa = wmma_bf16(a, frag_bf16(UtRow, kk + c0), sa);
    }
#pragma unroll
    for (int v = 0; v < 8; ++v) Sr[(m * 16 + v + 8 * hi) * 20 + ln] = sa[v];
    __syncthreads();

    // ---- A2: Hrec = Srec @ V_m (K=16 zero-padded to 32); H_in = tanh(P+rec+cb) ----
    v16bf a2 = fragA_k16_f32(Sr + (m * 16 + ln) * 20, c0);
#pragma unroll
    for (int ct = 0; ct < 8; ++ct) {
      v16bf bv = fragB_k16_bf16(Vt + (m * 128 + ct * 16 + ln) * 24, c0);
      v8f h8 = wmma_bf16(a2, bv, vzero8());
      int col = ct * 16 + ln;
      float cbv = cb[m * kH + col];
#pragma unroll
      for (int v = 0; v < 8; ++v) {
        int row = v + 8 * hi;
        float p = pc[((size_t)(t * kB + b0 + row) * kM + m) * kH + col];
        Hs[(m * 16 + row) * 132 + col] = tanhf(p + h8[v] + cbv);   // wave-local Hs[m]
      }
    }
    __syncthreads();

    // ---- A3: Kc/Qc = H_in_m @ W_K/W_Q  (16x16, K=128) ----
    v8f ka = vzero8(), qa = vzero8();
#pragma unroll
    for (int kk = 0; kk < 128; kk += 32) {
      v16bf a = frag_f32(HsRow, kk + c0);
      ka = wmma_bf16(a, frag_bf16(WKRow, kk + c0), ka);
      qa = wmma_bf16(a, frag_bf16(WQRow, kk + c0), qa);
    }
#pragma unroll
    for (int v = 0; v < 8; ++v) {
      Kc[(m * 16 + v + 8 * hi) * 16 + ln] = ka[v];
      Qc[(m * 16 + v + 8 * hi) * 16 + ln] = qa[v];
    }
    __syncthreads();

    // ---- B: scores (Q.K / sqrt(16)) + softmax over n ----
    if (tid < 128) {
      int row = tid >> 3, mq = tid & 7;
      float sc[8];
#pragma unroll
      for (int n = 0; n < 8; ++n) {
        float s = 0.0f;
#pragma unroll
        for (int kd = 0; kd < 16; ++kd)
          s += Qc[(mq * 16 + row) * 16 + kd] * Kc[(n * 16 + row) * 16 + kd];
        sc[n] = s * 0.25f;
      }
      float mx = sc[0];
#pragma unroll
      for (int n = 1; n < 8; ++n) mx = fmaxf(mx, sc[n]);
      float den = 0.0f;
#pragma unroll
      for (int n = 0; n < 8; ++n) { sc[n] = __expf(sc[n] - mx); den += sc[n]; }
      float inv = 1.0f / den;
#pragma unroll
      for (int n = 0; n < 8; ++n) { sc[n] *= inv; atl[row * 64 + mq * 8 + n] = sc[n]; }
      size_t ab = ((size_t)(t * kB + b0 + row) * kM + mq) * kM;
      *reinterpret_cast<float4*>(attn_out + ab)     = make_float4(sc[0], sc[1], sc[2], sc[3]);
      *reinterpret_cast<float4*>(attn_out + ab + 4) = make_float4(sc[4], sc[5], sc[6], sc[7]);
    }
    __syncthreads();

    // ---- C: A_comm + H_new; thread owns (rowC, 4-wide h chunk) across ALL m,
    //         so its Hs reads/writes never overlap another thread's. ----
#pragma unroll
    for (int it = 0; it < 2; ++it) {
      int h0 = (qC + it * 16) * 4;
      float4 v4[8], ac[8];
#pragma unroll
      for (int mm = 0; mm < 8; ++mm)
        v4[mm] = *reinterpret_cast<const float4*>(Hs + (mm * 16 + rowC) * 132 + h0);
#pragma unroll
      for (int n = 0; n < 8; ++n) ac[n] = make_float4(0.f, 0.f, 0.f, 0.f);
#pragma unroll
      for (int mm = 0; mm < 8; ++mm) {
        float4 d4 = *reinterpret_cast<const float4*>(dVd + mm * kH + h0);
        float4 w4 = make_float4(v4[mm].x * d4.x, v4[mm].y * d4.y,
                                v4[mm].z * d4.z, v4[mm].w * d4.w);
        const float* arow2 = atl + rowC * 64 + mm * 8;
#pragma unroll
        for (int n = 0; n < 8; ++n) {
          float awt = arow2[n];
          ac[n].x += awt * w4.x; ac[n].y += awt * w4.y;
          ac[n].z += awt * w4.z; ac[n].w += awt * w4.w;
        }
      }
#pragma unroll
      for (int n = 0; n < 8; ++n) {
        float4 hn = make_float4(v4[n].x + ac[n].x, v4[n].y + ac[n].y,
                                v4[n].z + ac[n].z, v4[n].w + ac[n].w);
        *reinterpret_cast<float4*>(Hs + (n * 16 + rowC) * 132 + h0) = hn;
        size_t gb = ((size_t)(t * kB + b0 + rowC) * kM + n) * kH + h0;
        *reinterpret_cast<float4*>(hid_out + gb) = hn;
        *reinterpret_cast<float4*>(pc + gb) = ac[n];   // A_comm overwrites input_part[t]
      }
    }
    if (t + 1 < kS) {  // prefetch next step's input_part slice (global_prefetch_b8)
      const float* pf = pc + ((size_t)((t + 1) * kB + b0) * kM) * kH + tid * 64;
      __builtin_prefetch(pf, 0, 1);
      __builtin_prefetch(pf + 32, 0, 1);
    }
    __syncthreads();
  }
}

// ---------------------------------------------------------------------------
// k3: outputs[b,s,:] = hiddens[s,b,:] @ Wo^T + bo.  Wo staged bf16 in LDS.
// grid = 512 blocks x 8 waves, one 16-row tile per wave (4096 tiles total).
// ---------------------------------------------------------------------------
__global__ void __launch_bounds__(256)
k3_out_proj(const float* __restrict__ hid, const float* __restrict__ Wo,
            const float* __restrict__ bo, float* __restrict__ outp) {
  extern __shared__ __align__(16) char smem[];
  bf16*  Wob = reinterpret_cast<bf16*>(smem);             // [128][1032]
  float* bol = reinterpret_cast<float*>(smem + 264192);   // [128]
  const int tid = threadIdx.x;
#pragma unroll 1
  for (int idx = tid; idx < kO * (kM * kH); idx += 256) {
    int o = idx >> 10, k = idx & 1023;
    Wob[o * 1032 + k] = (bf16)Wo[idx];                    // Bt[n=o][k] = Wo[o][k]
  }
  if (tid < kO) bol[tid] = bo[tid];
  __syncthreads();

  const int w = tid >> 5, lane = tid & 31, ln = lane & 15, hi = lane >> 4, c0 = hi << 3;
  const int tl = blockIdx.x * 8 + w;
  const float* arow = hid + (size_t)(tl * 16 + ln) * (kM * kH);

  v8f acc[8];
#pragma unroll
  for (int ct = 0; ct < 8; ++ct) acc[ct] = vzero8();
#pragma unroll 4
  for (int kk = 0; kk < 1024; kk += 32) {
    v16bf a = frag_f32(arow, kk + c0);
#pragma unroll
    for (int ct = 0; ct < 8; ++ct)
      acc[ct] = wmma_bf16(a, frag_bf16(Wob + (ct * 16 + ln) * 1032, kk + c0), acc[ct]);
  }
#pragma unroll
  for (int ct = 0; ct < 8; ++ct) {
    int col = ct * 16 + ln;
    float bb = bol[col];
#pragma unroll
    for (int v = 0; v < 8; ++v) {
      int row = tl * 16 + v + 8 * hi;            // row = t*B + b
      int tt = row >> 7, bbx = row & 127;
      outp[((size_t)bbx * kS + tt) * kO + col] = acc[ct][v] + bb;
    }
  }
}

// ---------------------------------------------------------------------------
extern "C" void kernel_launch(void* const* d_in, const int* in_sizes, int n_in,
                              void* d_out, int out_size, void* d_ws, size_t ws_size,
                              hipStream_t stream) {
  (void)in_sizes; (void)n_in; (void)out_size; (void)d_ws; (void)ws_size;
  const float* x    = (const float*)d_in[0];
  const float* Wp   = (const float*)d_in[1];
  const float* bp   = (const float*)d_in[2];
  const float* U    = (const float*)d_in[3];
  const float* V    = (const float*)d_in[4];
  const float* cbia = (const float*)d_in[5];
  const float* W_in = (const float*)d_in[6];
  const float* W_Q  = (const float*)d_in[7];
  const float* W_K  = (const float*)d_in[8];
  const float* W_V  = (const float*)d_in[9];
  const float* Wo   = (const float*)d_in[10];
  const float* bo   = (const float*)d_in[11];

  float* out     = (float*)d_out;
  float* outputs = out + OUT_OUT;
  float* hid     = out + OUT_HID;
  float* ain     = out + OUT_AIN;
  float* acomm   = out + OUT_ACOMM;   // scratch for input_part, then A_comm
  float* attn    = out + OUT_ATTN;

  k1_input_proj<<<dim3(4096), dim3(256), 137728, stream>>>(x, Wp, bp, W_in, ain, acomm);
  k2_recurrence<<<dim3(8), dim3(256), 260096, stream>>>(U, V, cbia, W_Q, W_K, W_V,
                                                        acomm, hid, attn);
  k3_out_proj<<<dim3(512), dim3(256), 264704, stream>>>(hid, Wo, bo, outputs);
}
